// Attn_10445360463906
// MI455X (gfx1250) — matrix operand order repaired and checked by reading, not compiled, not run on hardware
//
#include <hip/hip_runtime.h>
#include <hip/hip_bf16.h>
#include <math.h>
#include <stdint.h>

typedef __attribute__((ext_vector_type(16))) __bf16 v16bf;
typedef __attribute__((ext_vector_type(8)))  float  v8f;

union FragBF { v16bf v; uint4 q[2]; };

#define WMMA_BF16(A,B,C) \
  __builtin_amdgcn_wmma_f32_16x16x32_bf16(false,(A),false,(B),(short)0,(C),false,false)

__device__ __forceinline__ float rowmax16(float x){
  x = fmaxf(x, __shfl_xor(x, 1, 32));
  x = fmaxf(x, __shfl_xor(x, 2, 32));
  x = fmaxf(x, __shfl_xor(x, 4, 32));
  x = fmaxf(x, __shfl_xor(x, 8, 32));
  return x;
}
__device__ __forceinline__ float rowsum16(float x){
  x += __shfl_xor(x, 1, 32);
  x += __shfl_xor(x, 2, 32);
  x += __shfl_xor(x, 4, 32);
  x += __shfl_xor(x, 8, 32);
  return x;
}

// ---------------------------------------------------------------- conversions
__global__ void cvt_f32_to_bf16(const float* __restrict__ in,
                                __bf16* __restrict__ out, long n){
  long i = (long)blockIdx.x * blockDim.x + threadIdx.x;
  long stride = (long)gridDim.x * blockDim.x;
  for (; i < n; i += stride) out[i] = (__bf16)in[i];
}

// W is [Kd][Nd] row-major f32; Wt is [Nd][Kd] row-major bf16.
__global__ void transpose_cvt(const float* __restrict__ W,
                              __bf16* __restrict__ Wt, int Kd, int Nd){
  long total = (long)Kd * Nd;
  long i = (long)blockIdx.x * blockDim.x + threadIdx.x;
  long stride = (long)gridDim.x * blockDim.x;
  for (; i < total; i += stride){
    int  k = (int)(i % Kd);
    long n = i / Kd;
    Wt[i] = (__bf16)W[(long)k * Nd + n];
  }
}

// ---------------------------------------------------------------- GEMM (WMMA)
// C[M][N] = A[M][K] * Bt[N][K]^T + bias.  128x128 block tile, BK=32,
// double-buffered LDS staged via GLOBAL_LOAD_ASYNC_TO_LDS_B128 (ASYNCcnt),
// 4 waves (2x2), each wave computes a 64x64 tile = 16 WMMA per K-step.
__global__ __launch_bounds__(128) void gemm_bf16_wmma(
    const __bf16* __restrict__ A,    // [M][Kd]
    const __bf16* __restrict__ Bt,   // [N][Kd]
    const float*  __restrict__ bias, // [N]
    __bf16* __restrict__ outBf,      // [M][N] or nullptr
    float*  __restrict__ outF,       // [M][N] or nullptr
    int M, int N, int Kd)
{
  __shared__ __align__(16) __bf16 sA[2][128][32];
  __shared__ __align__(16) __bf16 sB[2][128][32];

  const int tid  = threadIdx.x;
  const int wave = tid >> 5, lane = tid & 31;
  const int hi   = lane >> 4, l16 = lane & 15;
  const int wm   = wave & 1,  wn  = wave >> 1;
  const int rowBase = blockIdx.y * 128;
  const int colBase = blockIdx.x * 128;

  v8f acc[4][4] = {};

  // Async memory->LDS staging: no VGPR round-trip, tracked with ASYNCcnt.
  auto stage = [&](int buf, int k0){
    #pragma unroll
    for (int it = 0; it < 4; ++it){
      int v = tid + it * 128;          // 0..511 : 128 rows x 4 vec16B
      int r = v >> 2, c = (v & 3) * 8;
      unsigned la = (unsigned)(uintptr_t)&sA[buf][r][c];
      unsigned lb = (unsigned)(uintptr_t)&sB[buf][r][c];
      unsigned oa = ((unsigned)(rowBase + r) * (unsigned)Kd + (unsigned)(k0 + c)) * 2u;
      unsigned ob = ((unsigned)(colBase + r) * (unsigned)Kd + (unsigned)(k0 + c)) * 2u;
      asm volatile("global_load_async_to_lds_b128 %0, %1, %2"
                   :: "v"(la), "v"(oa), "s"(A) : "memory");
      asm volatile("global_load_async_to_lds_b128 %0, %1, %2"
                   :: "v"(lb), "v"(ob), "s"(Bt) : "memory");
    }
  };

  stage(0, 0);
  asm volatile("s_wait_asynccnt 0x0" ::: "memory");
  __syncthreads();

  const int nk = Kd >> 5;
  int buf = 0;
  for (int kt = 0; kt < nk; ++kt){
    if (kt + 1 < nk){
      __builtin_prefetch(&A[(size_t)(rowBase + tid) * Kd + (kt + 1) * 32], 0, 3);
      stage(buf ^ 1, (kt + 1) * 32);   // overlaps with the 16 WMMAs below
    }
    FragBF af[4], bfr[4];
    #pragma unroll
    for (int mt = 0; mt < 4; ++mt){
      int r = wm * 64 + mt * 16 + l16;           // A layout: 16 contiguous K per lane half
      af[mt].q[0] = *(const uint4*)&sA[buf][r][hi * 16];
      af[mt].q[1] = *(const uint4*)&sA[buf][r][hi * 16 + 8];
    }
    #pragma unroll
    for (int nt = 0; nt < 4; ++nt){
      int n = wn * 64 + nt * 16 + l16;           // B layout: 16 contiguous K per lane half
      bfr[nt].q[0] = *(const uint4*)&sB[buf][n][hi * 16];
      bfr[nt].q[1] = *(const uint4*)&sB[buf][n][hi * 16 + 8];
    }
    #pragma unroll
    for (int mt = 0; mt < 4; ++mt)
      #pragma unroll
      for (int nt = 0; nt < 4; ++nt)
        acc[mt][nt] = WMMA_BF16(af[mt].v, bfr[nt].v, acc[mt][nt]);
    asm volatile("s_wait_asynccnt 0x0" ::: "memory");
    __syncthreads();
    buf ^= 1;
  }

  #pragma unroll
  for (int mt = 0; mt < 4; ++mt){
    #pragma unroll
    for (int nt = 0; nt < 4; ++nt){
      int c = colBase + wn * 64 + nt * 16 + l16;
      float bv = bias ? bias[c] : 0.f;
      #pragma unroll
      for (int v = 0; v < 8; ++v){
        int r = rowBase + wm * 64 + mt * 16 + v + 8 * hi;   // C layout: row = v + 8*half
        float val = acc[mt][nt][v] + bv;
        if (outBf) outBf[(size_t)r * N + c] = (__bf16)val;
        else       outF [(size_t)r * N + c] = val;
      }
    }
  }
}

// ------------------------------------------------- RMSNorm + 2D RoPE + V-transpose
// qkv: [B*N][3*D] bf16, col = which*D + h*64 + d.
// Q,K out: [B][H][N][64] bf16.  Vt out: [B][H][64][N] bf16.
__global__ __launch_bounds__(256) void qkv_postproc(
    const __bf16* __restrict__ qkv,
    const float* __restrict__ q_scale, const float* __restrict__ k_scale,
    __bf16* __restrict__ Qo, __bf16* __restrict__ Ko, __bf16* __restrict__ Vt)
{
  const int H = 16, D = 1024, Nn = 1024;
  long tid = (long)blockIdx.x * blockDim.x + threadIdx.x;
  long rows = 8L * Nn * H;
  if (tid >= rows) return;
  int  h  = (int)(tid & 15);
  long bn = tid >> 4;                 // b*N + n
  int  n  = (int)(bn & 1023);
  long b  = bn >> 10;
  size_t base   = (size_t)bn * (3 * D);
  size_t outRow = (((size_t)b * H + h) * Nn + n) * 64;

  // axial RoPE tables: angles = concat(rowIdx*invf, colIdx*invf), dq=16
  float cosv[32], sinv[32];
  int rI = n >> 5, cI = n & 31;       // side = 32
  #pragma unroll
  for (int j = 0; j < 16; ++j){
    float invf = __expf(-(float)j * (9.210340371976184f / 16.f)); // theta^{-j/16}
    float ar = rI * invf, ac = cI * invf;
    cosv[j]      = __cosf(ar); sinv[j]      = __sinf(ar);
    cosv[16 + j] = __cosf(ac); sinv[16 + j] = __sinf(ac);
  }

  float t[64];
  // ---- Q
  float ss = 0.f;
  #pragma unroll
  for (int d = 0; d < 64; ++d){ t[d] = (float)qkv[base + h * 64 + d]; ss += t[d] * t[d]; }
  float rinv = rsqrtf(ss * (1.f / 64.f) + 1e-6f);
  #pragma unroll
  for (int d = 0; d < 64; ++d) t[d] *= rinv * q_scale[d];
  #pragma unroll
  for (int d = 0; d < 32; ++d){
    float t1 = t[d], t2 = t[d + 32];
    Qo[outRow + d]      = (__bf16)(t1 * cosv[d] - t2 * sinv[d]);
    Qo[outRow + 32 + d] = (__bf16)(t1 * sinv[d] + t2 * cosv[d]);
  }
  // ---- K
  ss = 0.f;
  #pragma unroll
  for (int d = 0; d < 64; ++d){ t[d] = (float)qkv[base + D + h * 64 + d]; ss += t[d] * t[d]; }
  rinv = rsqrtf(ss * (1.f / 64.f) + 1e-6f);
  #pragma unroll
  for (int d = 0; d < 64; ++d) t[d] *= rinv * k_scale[d];
  #pragma unroll
  for (int d = 0; d < 32; ++d){
    float t1 = t[d], t2 = t[d + 32];
    Ko[outRow + d]      = (__bf16)(t1 * cosv[d] - t2 * sinv[d]);
    Ko[outRow + 32 + d] = (__bf16)(t1 * sinv[d] + t2 * cosv[d]);
  }
  // ---- V (transpose to [dim][key] so PV B-fragments are K-contiguous)
  size_t vtBase = ((size_t)b * H + h) * 64;
  #pragma unroll
  for (int d = 0; d < 64; ++d)
    Vt[(vtBase + d) * Nn + n] = qkv[base + 2 * D + h * 64 + d];
}

// ---------------------------------------------------------------- flash attention
// grid: (N/64, B*H).  4 waves, each wave owns 16 q rows; keys streamed 32 at a time.
// O out: [B][N][1024] bf16 (col = h*64 + d) feeding the output-projection GEMM.
__global__ __launch_bounds__(128) void flash_attn(
    const __bf16* __restrict__ Q, const __bf16* __restrict__ Km,
    const __bf16* __restrict__ Vt, __bf16* __restrict__ O)
{
  const int Nn = 1024;
  __shared__ __align__(16) __bf16 sP[4][16][32];   // per-wave P tile (16 q x 32 k)

  int wave = threadIdx.x >> 5, lane = threadIdx.x & 31;
  int hi = lane >> 4, l16 = lane & 15;
  int bh = blockIdx.y;                // b*16 + h
  int b = bh >> 4, h = bh & 15;
  int qr0 = blockIdx.x * 64 + wave * 16;

  const __bf16* Qp = Q  + (size_t)bh * Nn * 64;
  const __bf16* Kp = Km + (size_t)bh * Nn * 64;
  const __bf16* Vp = Vt + (size_t)bh * 64 * Nn;

  FragBF aq[2];
  #pragma unroll
  for (int kk = 0; kk < 2; ++kk){
    const __bf16* src = &Qp[(size_t)(qr0 + l16) * 64 + kk * 32];
    aq[kk].q[0] = *(const uint4*)&src[hi * 16];
    aq[kk].q[1] = *(const uint4*)&src[hi * 16 + 8];
  }

  float m[8], l[8];
  #pragma unroll
  for (int v = 0; v < 8; ++v){ m[v] = -3.0e38f; l[v] = 0.f; }
  v8f o[4] = {};

  for (int kc = 0; kc < Nn; kc += 32){
    v8f s[2] = {};
    #pragma unroll
    for (int t = 0; t < 2; ++t){
      #pragma unroll
      for (int kk = 0; kk < 2; ++kk){
        FragBF kb;
        const __bf16* src = &Kp[(size_t)(kc + t * 16 + l16) * 64 + kk * 32 + hi * 16];
        kb.q[0] = *(const uint4*)&src[0];
        kb.q[1] = *(const uint4*)&src[8];
        s[t] = WMMA_BF16(aq[kk].v, kb.v, s[t]);
      }
    }
    const float scale = 0.125f;       // 64^-0.5
    float p[2][8];
    #pragma unroll
    for (int v = 0; v < 8; ++v){
      float a = rowmax16(fmaxf(s[0][v], s[1][v]) * scale);
      float mn = fmaxf(m[v], a);
      float corr = __expf(m[v] - mn);
      m[v] = mn;
      float p0 = __expf(s[0][v] * scale - mn);
      float p1 = __expf(s[1][v] * scale - mn);
      p[0][v] = p0; p[1][v] = p1;
      l[v] = l[v] * corr + rowsum16(p0 + p1);
      #pragma unroll
      for (int d = 0; d < 4; ++d) o[d][v] = o[d][v] * corr;
    }
    // re-layout P (C layout -> A fragment) through wave-private LDS
    #pragma unroll
    for (int t = 0; t < 2; ++t)
      #pragma unroll
      for (int v = 0; v < 8; ++v)
        sP[wave][v + 8 * hi][t * 16 + l16] = (__bf16)p[t][v];
    asm volatile("s_wait_dscnt 0x0" ::: "memory");   // wave-local, no block barrier
    FragBF pa;
    pa.q[0] = *(const uint4*)&sP[wave][l16][hi * 16];
    pa.q[1] = *(const uint4*)&sP[wave][l16][hi * 16 + 8];
    #pragma unroll
    for (int d = 0; d < 4; ++d){
      FragBF vb;
      const __bf16* src = &Vp[(size_t)(d * 16 + l16) * Nn + kc + hi * 16];
      vb.q[0] = *(const uint4*)&src[0];
      vb.q[1] = *(const uint4*)&src[8];
      o[d] = WMMA_BF16(pa.v, vb.v, o[d]);
    }
  }

  #pragma unroll
  for (int d = 0; d < 4; ++d)
    #pragma unroll
    for (int v = 0; v < 8; ++v){
      int r = qr0 + v + 8 * hi;
      float val = o[d][v] / l[v];
      O[((size_t)b * Nn + r) * 1024 + h * 64 + d * 16 + l16] = (__bf16)val;
    }
}

// ---------------------------------------------------------------- launcher
extern "C" void kernel_launch(void* const* d_in, const int* in_sizes, int n_in,
                              void* d_out, int out_size, void* d_ws, size_t ws_size,
                              hipStream_t stream)
{
  const float* x       = (const float*)d_in[0];
  const float* Wqkv    = (const float*)d_in[1];
  const float* bqkv    = (const float*)d_in[2];
  const float* Wout    = (const float*)d_in[3];
  const float* bout    = (const float*)d_in[4];
  const float* q_scale = (const float*)d_in[5];
  const float* k_scale = (const float*)d_in[6];
  float* out = (float*)d_out;

  const long M = 8L * 1024;   // B*N
  const int  D = 1024;

  char* w = (char*)d_ws;
  __bf16* xb    = (__bf16*)w;  w += (size_t)M * D * 2;          // 16 MB
  __bf16* wqkvt = (__bf16*)w;  w += (size_t)3 * D * D * 2;      //  6 MB
  __bf16* woutt = (__bf16*)w;  w += (size_t)D * D * 2;          //  2 MB
  __bf16* qkvb  = (__bf16*)w;  w += (size_t)M * 3 * D * 2;      // 48 MB
  __bf16* Qb    = (__bf16*)w;  w += (size_t)M * D * 2;          // 16 MB
  __bf16* Kb    = (__bf16*)w;  w += (size_t)M * D * 2;          // 16 MB
  __bf16* Vtb   = (__bf16*)w;  w += (size_t)M * D * 2;          // 16 MB
  __bf16* Ob    = (__bf16*)w;  w += (size_t)M * D * 2;          // 16 MB

  cvt_f32_to_bf16<<<4096, 256, 0, stream>>>(x, xb, M * D);
  transpose_cvt  <<<4096, 256, 0, stream>>>(Wqkv, wqkvt, D, 3 * D);
  transpose_cvt  <<<2048, 256, 0, stream>>>(Wout, woutt, D, D);

  gemm_bf16_wmma<<<dim3(24, 64), 128, 0, stream>>>(
      xb, wqkvt, bqkv, qkvb, nullptr, (int)M, 3 * D, D);

  qkv_postproc<<<512, 256, 0, stream>>>(qkvb, q_scale, k_scale, Qb, Kb, Vtb);

  flash_attn<<<dim3(16, 128), 128, 0, stream>>>(Qb, Kb, Vtb, Ob);

  gemm_bf16_wmma<<<dim3(8, 64), 128, 0, stream>>>(
      Ob, woutt, bout, nullptr, out, (int)M, D, D);
}
